// MultiHeadAttension_9990093930612
// MI455X (gfx1250) — compile-verified, matrix operand order checked by reading, not run on hardware
//
#include <hip/hip_runtime.h>

// ---------------------------------------------------------------------------
// MI455X (gfx1250) multi-head attention block: bf16 WMMA + async LDS staging.
// ---------------------------------------------------------------------------

typedef __bf16 bf16_t;
typedef __attribute__((ext_vector_type(16))) __bf16 v16bf;
typedef __attribute__((ext_vector_type(8)))  __bf16 v8bf;
typedef __attribute__((ext_vector_type(8)))  float  v8f;
typedef __attribute__((ext_vector_type(4)))  int    v4i;

#define B_  4
#define L_  2048
#define E_  1024
#define H_  16
#define HS_ 64

#if defined(__gfx1250__) && __has_builtin(__builtin_amdgcn_global_load_async_to_lds_b128)
#define ASYNC_LDS 1
typedef __attribute__((address_space(1))) v4i gv4i_t;   // global v4i
typedef __attribute__((address_space(3))) v4i lv4i_t;   // LDS v4i
#endif

__device__ __forceinline__ bf16_t f2bf(float f) {
  union { float f; unsigned u; } c; c.f = f;
  unsigned u = c.u;
  u += 0x7FFFu + ((u >> 16) & 1u);   // round-to-nearest-even
  union { unsigned short s; bf16_t b; } o;
  o.s = (unsigned short)(u >> 16);
  return o.b;
}

// Per-lane fragment slice for 16x16x32 bf16 WMMA: lane half selects k-offset
// {0,8}; values are p[0..7] and p[16..23] relative to (row base + koff).
__device__ __forceinline__ v16bf load_frag(const bf16_t* p) {
  v8bf lo = *(const v8bf*)(p);
  v8bf hi = *(const v8bf*)(p + 16);
  v16bf r;
#pragma unroll
  for (int i = 0; i < 8; ++i) { r[i] = lo[i]; r[i + 8] = hi[i]; }
  return r;
}

__device__ __forceinline__ v8f wmma_bf16(v16bf a, v16bf b, v8f c) {
  return __builtin_amdgcn_wmma_f32_16x16x32_bf16(false, a, false, b,
                                                 (short)0, c, false, false);
}

// 16-byte global -> LDS copy. Async (ASYNCcnt-tracked, no VGPR transit) when
// the gfx1250 builtin is available; plain copy otherwise.
__device__ __forceinline__ void cp16(bf16_t* lds_dst, const bf16_t* gsrc) {
#ifdef ASYNC_LDS
  __builtin_amdgcn_global_load_async_to_lds_b128(
      (gv4i_t*)gsrc, (lv4i_t*)lds_dst, 0, 0);
#else
  *(v8bf*)lds_dst = *(const v8bf*)gsrc;
#endif
}

__device__ __forceinline__ void cp_wait() {
#ifdef ASYNC_LDS
  asm volatile("s_wait_asynccnt 0" ::: "memory");
#endif
}

// Wave-local LDS ordering for the P transpose (DS ops are in-order per wave).
__device__ __forceinline__ void wave_lds_fence() {
  __builtin_amdgcn_wave_barrier();
  asm volatile("s_wait_dscnt 0" ::: "memory");
  __builtin_amdgcn_wave_barrier();
}

// ---------------------------------------------------------------------------
// 1) elementwise fp32 -> bf16
// ---------------------------------------------------------------------------
__global__ __launch_bounds__(256) void cvt_bf16(const float* __restrict__ in,
                                                bf16_t* __restrict__ out, int n) {
  int i = blockIdx.x * blockDim.x + threadIdx.x;
  if (i < n) out[i] = f2bf(in[i]);
}

// 2) transpose-convert weights: W[K][N] fp32 -> WT[N][K] bf16
__global__ __launch_bounds__(256) void cvt_bf16_tr(const float* __restrict__ in,
                                                   bf16_t* __restrict__ out,
                                                   int K, int N) {
  int i = blockIdx.x * blockDim.x + threadIdx.x;
  if (i < N * K) {
    int n = i / K, k = i - n * K;
    out[i] = f2bf(in[(size_t)k * N + n]);
  }
}

// ---------------------------------------------------------------------------
// 3) QKV GEMM. Block = 8 waves, 128(M) x 64(N) tile. The 64xK B strip is
//    shared by all 8 waves -> cooperatively staged in double-buffered LDS via
//    async copies; A fragments register-double-buffered.
//    Epilogue scatters into K,Q [B,H,L,HS] and V transposed [B,H,HS,L].
//    Reference split order along N is (k, q, v).
// ---------------------------------------------------------------------------
__global__ __launch_bounds__(256) void qkv_gemm(const bf16_t* __restrict__ X,
                                                const bf16_t* __restrict__ WT,
                                                bf16_t* __restrict__ Kt,
                                                bf16_t* __restrict__ Qt,
                                                bf16_t* __restrict__ Vt) {
  __shared__ bf16_t Bs[2][64 * 32];          // [n][k] tile, double-buffered

  const int tid  = threadIdx.x;
  const int lane = tid & 31;
  const int wave = tid >> 5;
  const int l15  = lane & 15;
  const int hl   = lane >> 4;
  const int koff = hl * 8;
  const int m0   = blockIdx.x * 128 + wave * 16;
  const int n0   = blockIdx.y * 64;

  const bf16_t* arow = X + (size_t)(m0 + l15) * E_ + koff;

  // stage B tile for k-window [kc*32, kc*32+32): 256 x 16B chunks, 1/thread
  const int srow  = tid >> 2;                 // 0..63
  const int spart = tid & 3;                  // 0..3
  const bf16_t* bsrc = WT + (size_t)(n0 + srow) * E_ + spart * 8;

  cp16(&Bs[0][srow * 32 + spart * 8], bsrc);           // stage kc=0
  v16bf a_cur = load_frag(arow);                       // A frag kc=0
  cp_wait();
  __syncthreads();

  v8f acc[4] = {};
  const int KSTEPS = E_ / 32;
  for (int kc = 0; kc < KSTEPS; ++kc) {
    const int cur = kc & 1;
    v16bf a_nxt = a_cur;
    if (kc + 1 < KSTEPS) {
      cp16(&Bs[(kc + 1) & 1][srow * 32 + spart * 8], bsrc + (kc + 1) * 32);
      __builtin_prefetch(arow + (kc + 2) * 32, 0, 1);
      a_nxt = load_frag(arow + (kc + 1) * 32);
    }
#pragma unroll
    for (int t = 0; t < 4; ++t) {
      v16bf b = load_frag(&Bs[cur][(t * 16 + l15) * 32 + koff]);
      acc[t] = wmma_bf16(a_cur, b, acc[t]);
    }
    a_cur = a_nxt;
    if (kc + 1 < KSTEPS) cp_wait();
    __syncthreads();
  }

  // C/D layout: lane holds column n=(lane&15)+16*tile, rows m = v + 8*(lane>=16)
#pragma unroll
  for (int t = 0; t < 4; ++t) {
    int n   = n0 + t * 16 + l15;
    int sec = n >> 10;        // 0:k 1:q 2:v
    int e   = n & 1023;
    int h   = e >> 6;
    int d   = e & 63;
#pragma unroll
    for (int v = 0; v < 8; ++v) {
      int m = m0 + v + 8 * hl;
      int b = m >> 11;
      int l = m & 2047;
      bf16_t val = f2bf(acc[t][v]);
      size_t bh = (size_t)b * H_ + h;
      if (sec == 0)      Kt[(bh * L_ + l) * HS_ + d] = val;
      else if (sec == 1) Qt[(bh * L_ + l) * HS_ + d] = val;
      else               Vt[(bh * HS_ + d) * L_ + l] = val;
    }
  }
}

// ---------------------------------------------------------------------------
// 4) Causal attention, flash-style online softmax.
//    Block = 4 waves covering 64 rows of one (b,h); wave owns 16 rows.
//    Q (32x64) and V (64x32) chunks staged cooperatively in double-buffered
//    LDS (shared by all 4 waves), block-uniform causal trip count.
// ---------------------------------------------------------------------------
__global__ __launch_bounds__(128) void attn_kernel(const bf16_t* __restrict__ Kt,
                                                   const bf16_t* __restrict__ Qt,
                                                   const bf16_t* __restrict__ Vt,
                                                   bf16_t* __restrict__ Y) {
  __shared__ bf16_t Qs[2][32 * 64];     // q rows jb..jb+31, contiguous copy
  __shared__ bf16_t Vs[2][64 * 32];     // Vt rows d=0..63, cols jb..jb+31
  __shared__ bf16_t pbuf[4][16 * 32];   // per-wave P staging (C -> A layout)

  const int tid  = threadIdx.x;
  const int lane = tid & 31;
  const int wave = tid >> 5;
  const int l15  = lane & 15;
  const int hl   = lane >> 4;
  const int koff = hl * 8;

  const int bh = blockIdx.y;
  const int b  = bh >> 4;
  const int h  = bh & 15;
  const bf16_t* Kp = Kt + (size_t)bh * L_ * HS_;
  const bf16_t* Qp = Qt + (size_t)bh * L_ * HS_;
  const bf16_t* Vp = Vt + (size_t)bh * HS_ * L_;

  const int i0b = blockIdx.x * 64;
  const int i0  = i0b + wave * 16;
  const int nch = (i0b + 64) >> 5;       // block-uniform causal j-chunks

  // A fragments: rows k_{i0+l15}, d-chunks [0,32) and [32,64)
  v16bf ka0 = load_frag(Kp + (size_t)(i0 + l15) * HS_ + koff);
  v16bf ka1 = load_frag(Kp + (size_t)(i0 + l15) * HS_ + 32 + koff);

  const int qc0 = tid, qc1 = tid + 128;            // Q chunks 0..255

  v8f acc[4] = {};
  float rowM[8], rowL[8];
#pragma unroll
  for (int v = 0; v < 8; ++v) { rowM[v] = -1e30f; rowL[v] = 0.f; }

  const float sc    = 0.125f;            // 1/sqrt(HS)
  const float log2e = 1.4426950408889634f;

  // ---- stage chunk 0 ----
  {
    cp16(&Qs[0][qc0 * 8], Qp + (size_t)qc0 * 8);
    cp16(&Qs[0][qc1 * 8], Qp + (size_t)qc1 * 8);
#pragma unroll
    for (int r = 0; r < 2; ++r) {
      int c = tid + r * 128;             // 0..255
      int row = c >> 2, part = c & 3;
      cp16(&Vs[0][row * 32 + part * 8], Vp + (size_t)row * L_ + part * 8);
    }
  }
  cp_wait();
  __syncthreads();

  for (int jc = 0; jc < nch; ++jc) {
    const int cur = jc & 1;
    const int jb  = jc * 32;

    // ---- stage chunk jc+1 (overlapped with compute) ----
    if (jc + 1 < nch) {
      const int nb = (jc + 1) & 1;
      const int jn = jb + 32;
      cp16(&Qs[nb][qc0 * 8], Qp + (size_t)jn * HS_ + qc0 * 8);
      cp16(&Qs[nb][qc1 * 8], Qp + (size_t)jn * HS_ + qc1 * 8);
#pragma unroll
      for (int r = 0; r < 2; ++r) {
        int c = tid + r * 128;
        int row = c >> 2, part = c & 3;
        cp16(&Vs[nb][row * 32 + part * 8], Vp + (size_t)row * L_ + jn + part * 8);
      }
    }

    // ---- S = K_tile x Q^T from LDS (two 16x16 tiles, K=64) ----
    v8f s0 = {}, s1 = {};
    s0 = wmma_bf16(ka0, load_frag(&Qs[cur][l15 * 64 + koff]),        s0);
    s0 = wmma_bf16(ka1, load_frag(&Qs[cur][l15 * 64 + 32 + koff]),   s0);
    s1 = wmma_bf16(ka0, load_frag(&Qs[cur][(16 + l15) * 64 + koff]), s1);
    s1 = wmma_bf16(ka1, load_frag(&Qs[cur][(16 + l15) * 64 + 32 + koff]), s1);

    const int j0 = jb + l15;
    const int j1 = jb + 16 + l15;

    float p0[8], p1[8];
#pragma unroll
    for (int v = 0; v < 8; ++v) {
      const int r = i0 + v + 8 * hl;
      float a = (j0 <= r) ? s0[v] * sc : -1e30f;
      float c = (j1 <= r) ? s1[v] * sc : -1e30f;
      float rmax = fmaxf(a, c);
#pragma unroll
      for (int m = 8; m; m >>= 1) rmax = fmaxf(rmax, __shfl_xor(rmax, m));
      float nm    = fmaxf(rowM[v], rmax);
      float scale = exp2f((rowM[v] - nm) * log2e);
      rowM[v] = nm;
      float e0 = (j0 <= r) ? exp2f((a - nm) * log2e) : 0.f;
      float e1 = (j1 <= r) ? exp2f((c - nm) * log2e) : 0.f;
      p0[v] = e0; p1[v] = e1;
      float rs = e0 + e1;
#pragma unroll
      for (int m = 8; m; m >>= 1) rs += __shfl_xor(rs, m);
      rowL[v] = rowL[v] * scale + rs;
#pragma unroll
      for (int t = 0; t < 4; ++t) acc[t][v] *= scale;
    }

    // ---- C-layout P -> per-wave LDS -> reload as A fragment ----
#pragma unroll
    for (int v = 0; v < 8; ++v) {
      int row = v + 8 * hl;
      pbuf[wave][row * 32 + l15]      = f2bf(p0[v]);
      pbuf[wave][row * 32 + 16 + l15] = f2bf(p1[v]);
    }
    wave_lds_fence();
    v16bf pa = load_frag(&pbuf[wave][l15 * 32 + koff]);
    wave_lds_fence();

    // ---- O += P (16x32) @ V (32x64) from LDS ----
#pragma unroll
    for (int t = 0; t < 4; ++t) {
      v16bf bv = load_frag(&Vs[cur][(t * 16 + l15) * 32 + koff]);
      acc[t] = wmma_bf16(pa, bv, acc[t]);
    }

    if (jc + 1 < nch) cp_wait();
    __syncthreads();
  }

  // normalize and write Y[b][l][h*HS+d] as bf16 for the FFN stage
#pragma unroll
  for (int v = 0; v < 8; ++v) {
    const int r   = i0 + v + 8 * hl;
    const float inv = 1.0f / rowL[v];
#pragma unroll
    for (int t = 0; t < 4; ++t) {
      int d = t * 16 + l15;
      Y[((size_t)b * L_ + r) * E_ + h * HS_ + d] = f2bf(acc[t][v] * inv);
    }
  }
}

// ---------------------------------------------------------------------------
// 5) FFN GEMM: Out[m][n] = sum_k Y[m][k] * WfT[n][k], fp32 output.
//    Same LDS-staged structure as qkv_gemm.
// ---------------------------------------------------------------------------
__global__ __launch_bounds__(256) void ffn_gemm(const bf16_t* __restrict__ Yb,
                                                const bf16_t* __restrict__ WT,
                                                float* __restrict__ Out) {
  __shared__ bf16_t Bs[2][64 * 32];

  const int tid  = threadIdx.x;
  const int lane = tid & 31;
  const int wave = tid >> 5;
  const int l15  = lane & 15;
  const int hl   = lane >> 4;
  const int koff = hl * 8;
  const int m0   = blockIdx.x * 128 + wave * 16;
  const int n0   = blockIdx.y * 64;

  const bf16_t* arow = Yb + (size_t)(m0 + l15) * E_ + koff;
  const int srow  = tid >> 2;
  const int spart = tid & 3;
  const bf16_t* bsrc = WT + (size_t)(n0 + srow) * E_ + spart * 8;

  cp16(&Bs[0][srow * 32 + spart * 8], bsrc);
  v16bf a_cur = load_frag(arow);
  cp_wait();
  __syncthreads();

  v8f acc[4] = {};
  const int KSTEPS = E_ / 32;
  for (int kc = 0; kc < KSTEPS; ++kc) {
    const int cur = kc & 1;
    v16bf a_nxt = a_cur;
    if (kc + 1 < KSTEPS) {
      cp16(&Bs[(kc + 1) & 1][srow * 32 + spart * 8], bsrc + (kc + 1) * 32);
      __builtin_prefetch(arow + (kc + 2) * 32, 0, 1);
      a_nxt = load_frag(arow + (kc + 1) * 32);
    }
#pragma unroll
    for (int t = 0; t < 4; ++t) {
      v16bf b = load_frag(&Bs[cur][(t * 16 + l15) * 32 + koff]);
      acc[t] = wmma_bf16(a_cur, b, acc[t]);
    }
    a_cur = a_nxt;
    if (kc + 1 < KSTEPS) cp_wait();
    __syncthreads();
  }

#pragma unroll
  for (int t = 0; t < 4; ++t) {
    int n = n0 + t * 16 + l15;
#pragma unroll
    for (int v = 0; v < 8; ++v) {
      int m = m0 + v + 8 * hl;
      Out[(size_t)m * E_ + n] = acc[t][v];
    }
  }
}

// ---------------------------------------------------------------------------
extern "C" void kernel_launch(void* const* d_in, const int* in_sizes, int n_in,
                              void* d_out, int out_size, void* d_ws, size_t ws_size,
                              hipStream_t stream) {
  (void)in_sizes; (void)n_in; (void)out_size; (void)ws_size;
  const float* x  = (const float*)d_in[0];   // [B,L,E]
  const float* Wa = (const float*)d_in[1];   // [E, 3E]
  const float* Wf = (const float*)d_in[2];   // [E, E]

  char* ws = (char*)d_ws;
  size_t off = 0;
  bf16_t* xb  = (bf16_t*)(ws + off); off += (size_t)B_ * L_ * E_ * 2;        // 16 MiB
  bf16_t* WaT = (bf16_t*)(ws + off); off += (size_t)3 * E_ * E_ * 2;         //  6 MiB
  bf16_t* WfT = (bf16_t*)(ws + off); off += (size_t)E_ * E_ * 2;             //  2 MiB
  bf16_t* Kt  = (bf16_t*)(ws + off); off += (size_t)B_ * H_ * L_ * HS_ * 2;  // 16 MiB
  bf16_t* Qt  = (bf16_t*)(ws + off); off += (size_t)B_ * H_ * L_ * HS_ * 2;  // 16 MiB
  bf16_t* Vt  = (bf16_t*)(ws + off); off += (size_t)B_ * H_ * L_ * HS_ * 2;  // 16 MiB
  bf16_t* Yb  = (bf16_t*)(ws + off); off += (size_t)B_ * L_ * E_ * 2;        // 16 MiB

  const int nx = B_ * L_ * E_;
  cvt_bf16<<<(nx + 255) / 256, 256, 0, stream>>>(x, xb, nx);

  const int nwa = E_ * 3 * E_;
  cvt_bf16_tr<<<(nwa + 255) / 256, 256, 0, stream>>>(Wa, WaT, E_, 3 * E_);

  const int nwf = E_ * E_;
  cvt_bf16_tr<<<(nwf + 255) / 256, 256, 0, stream>>>(Wf, WfT, E_, E_);

  qkv_gemm<<<dim3(64, 48), 256, 0, stream>>>(xb, WaT, Kt, Qt, Vt);
  attn_kernel<<<dim3(L_ / 64, B_ * H_), 128, 0, stream>>>(Kt, Qt, Vt, Yb);
  ffn_gemm<<<dim3(64, 16), 256, 0, stream>>>(Yb, WfT, (float*)d_out);
}
